// LLMMamba_11922829213847
// MI455X (gfx1250) — matrix operand order repaired and checked by reading, not compiled, not run on hardware
//
#include <hip/hip_runtime.h>

// ---------------------------------------------------------------------------
// Mamba 2-layer forward for MI455X (gfx1250, wave32).
// GEMMs run on v_wmma_f32_16x16x32_bf16 (bf16 in, f32 accum).
// LDS tiles are stored in WMMA *fragment order* so each lane reads its
// 32-byte fragment with contiguous ds_load_b128s (no scalar gather).
// ---------------------------------------------------------------------------

#define N_LAYERS 2
#define D_MODEL  768
#define D_INNER  1536
#define D_STATE  16
#define D_CONV   4
#define DT_RANK  48
#define VOCAB    50257
#define SEQ      1024
#define BATCH    2
#define MTOK     (BATCH * SEQ)
#define XDBL_W   (DT_RANK + 2 * D_STATE)   // 80

typedef __bf16 bf16_t;
typedef bf16_t v16bf __attribute__((ext_vector_type(16)));
typedef float  v8f   __attribute__((ext_vector_type(8)));

__device__ __forceinline__ bf16_t f2bf(float f) {
    unsigned u = __builtin_bit_cast(unsigned, f);
    unsigned r = (u + 0x7FFFu + ((u >> 16) & 1u)) >> 16;
    unsigned short s = (unsigned short)r;
    return __builtin_bit_cast(bf16_t, s);
}
__device__ __forceinline__ float silu_f(float x)    { return x / (1.0f + __expf(-x)); }
__device__ __forceinline__ float softplus_f(float x){ return (x > 20.0f) ? x : log1pf(__expf(x)); }

// ---------------------------------------------------------------------------
// Embedding gather: x[m, :] = emb[tokens[m], :]
// ---------------------------------------------------------------------------
__global__ void embed_kernel(const int* __restrict__ tokens,
                             const float* __restrict__ emb,
                             float* __restrict__ x) {
    int gid = blockIdx.x * blockDim.x + threadIdx.x;
    if (gid >= MTOK * D_MODEL) return;
    int m = gid / D_MODEL, c = gid % D_MODEL;
    x[gid] = emb[(size_t)tokens[m] * D_MODEL + c];
}

// ---------------------------------------------------------------------------
// RMSNorm -> bf16 (one block per row)
// ---------------------------------------------------------------------------
__global__ __launch_bounds__(256)
void rmsnorm_bf16_kernel(const float* __restrict__ x,
                         const float* __restrict__ w,
                         bf16_t* __restrict__ out, int D) {
    __shared__ float red[256];
    int m = blockIdx.x;
    const float* row = x + (size_t)m * D;
    float s = 0.0f;
    for (int c = threadIdx.x; c < D; c += 256) { float v = row[c]; s += v * v; }
    red[threadIdx.x] = s;
    __syncthreads();
    for (int off = 128; off > 0; off >>= 1) {
        if (threadIdx.x < off) red[threadIdx.x] += red[threadIdx.x + off];
        __syncthreads();
    }
    float rstd = rsqrtf(red[0] / (float)D + 1e-5f);
    for (int c = threadIdx.x; c < D; c += 256)
        out[(size_t)m * D + c] = f2bf(row[c] * rstd * w[c]);
}

// ---------------------------------------------------------------------------
// fp32 -> bf16 converter (for weights)
// ---------------------------------------------------------------------------
__global__ void cvt_bf16_kernel(const float* __restrict__ in,
                                bf16_t* __restrict__ out, int n) {
    int gid = blockIdx.x * blockDim.x + threadIdx.x;
    if (gid < n) out[gid] = f2bf(in[gid]);
}

// ---------------------------------------------------------------------------
// WMMA GEMM: C(MxN, f32) = A(MxK, bf16 row-major) * W(NxK, bf16 row-major)^T
// Block = 128 threads = 4 waves. Block tile 128(M) x 64(N), K stepped by 32.
// Wave w computes rows [w*32, w*32+32) x 64 cols -> 8 WMMA accumulators.
//
// LDS fragment-order layout (wave32; contiguous k, same order for A and B):
//   A 16x32 frag: lane m%16   (hi=0) holds K 0..15 contiguous in elems 0..15
//                 lane m%16+16(hi=1) holds K 16..31 contiguous
//   B 32x16 frag: lane n%16   holds K 0..15 contiguous
//                 lane n%16+16 holds K 16..31 contiguous
// Each lane's fragment is 32 contiguous bytes in LDS -> 2x ds_load_b128.
// ---------------------------------------------------------------------------
__global__ __launch_bounds__(128)
void gemm_bf16_wmma_kernel(const bf16_t* __restrict__ A,
                           const bf16_t* __restrict__ W,
                           float* __restrict__ C,
                           int M, int N, int K,
                           const float* __restrict__ residual) {
    // [subtile][lane][16 bf16] -- fragment order
    __shared__ __align__(32) bf16_t aF[8][32][16];   // 8 M-subtiles of 16 rows
    __shared__ __align__(32) bf16_t bF[4][32][16];   // 4 N-subtiles of 16 cols

    const int tid    = threadIdx.x;
    const int wave   = tid >> 5;
    const int lane   = tid & 31;
    const int lane16 = lane & 15;
    const int hi     = (lane >> 4) & 1;
    const int m0     = blockIdx.y * 128;
    const int n0     = blockIdx.x * 64;

    v8f acc[2][4];
    #pragma unroll
    for (int s = 0; s < 2; ++s)
        #pragma unroll
        for (int j = 0; j < 4; ++j)
            #pragma unroll
            for (int r = 0; r < 8; ++r) acc[s][j][r] = 0.0f;

    // A loader: thread t handles global row m0 + t, all 32 k of the step.
    const int a_row  = tid;            // 0..127
    const int a_wt   = a_row >> 4;     // M-subtile 0..7
    const int a_ll   = a_row & 15;     // lane-low within subtile
    // B loader: thread t handles W row n0 + (t>>1), k-half (t&1).
    const int b_n    = tid >> 1;       // 0..63
    const int b_half = tid & 1;        // 0: K0..15, 1: K16..31
    const int b_j    = b_n >> 4;       // N-subtile
    const int b_lane = (b_n & 15) + b_half * 16;

    for (int k0 = 0; k0 < K; k0 += 32) {
        __syncthreads();
        // ---- A tile: 128 rows x 32 k (M multiple of 128, K multiple of 32)
        {
            const uint4* src = (const uint4*)(A + (size_t)(m0 + a_row) * K + k0);
            uint4 q0 = src[0];   // K  0..7
            uint4 q1 = src[1];   // K  8..15
            uint4 q2 = src[2];   // K 16..23
            uint4 q3 = src[3];   // K 24..31
            *(uint4*)&aF[a_wt][a_ll     ][0] = q0;  // hi=0, elems 0..7
            *(uint4*)&aF[a_wt][a_ll + 16][0] = q2;  // hi=1, elems 0..7
            *(uint4*)&aF[a_wt][a_ll     ][8] = q1;  // hi=0, elems 8..15
            *(uint4*)&aF[a_wt][a_ll + 16][8] = q3;  // hi=1, elems 8..15
        }
        // ---- B tile: 64 rows (n) x 32 k, transposed role; guard n < N
        {
            uint4 s0, s1;
            if (n0 + b_n < N) {
                const uint4* src =
                    (const uint4*)(W + (size_t)(n0 + b_n) * K + k0 + b_half * 16);
                s0 = src[0];
                s1 = src[1];
            } else {
                s0 = make_uint4(0u, 0u, 0u, 0u);
                s1 = s0;
            }
            *(uint4*)&bF[b_j][b_lane][0] = s0;
            *(uint4*)&bF[b_j][b_lane][8] = s1;
        }
        if (k0 + 32 < K) {
            __builtin_prefetch(A + (size_t)(m0 + a_row) * K + k0 + 32, 0, 1);
            __builtin_prefetch(W + (size_t)(n0 + b_n) * K + k0 + 32, 0, 1);
        }
        __syncthreads();

        // ---- Compute: 2 A-frags x 4 B-frags = 8 WMMAs
        v16bf af0 = *(const v16bf*)&aF[wave * 2 + 0][lane][0];
        v16bf af1 = *(const v16bf*)&aF[wave * 2 + 1][lane][0];
        #pragma unroll
        for (int j = 0; j < 4; ++j) {
            v16bf bf = *(const v16bf*)&bF[j][lane][0];
            acc[0][j] = __builtin_amdgcn_wmma_f32_16x16x32_bf16(
                false, af0, false, bf, (short)0, acc[0][j], false, false);
            acc[1][j] = __builtin_amdgcn_wmma_f32_16x16x32_bf16(
                false, af1, false, bf, (short)0, acc[1][j], false, false);
        }
    }

    // ---- Store. C/D layout: VGPR r holds M = r + 8*hi, N = lane16.
    #pragma unroll
    for (int s = 0; s < 2; ++s) {
        #pragma unroll
        for (int j = 0; j < 4; ++j) {
            #pragma unroll
            for (int r = 0; r < 8; ++r) {
                int m = m0 + wave * 32 + s * 16 + hi * 8 + r;
                int n = n0 + j * 16 + lane16;
                if (n < N) {
                    size_t idx = (size_t)m * N + n;
                    float v = acc[s][j][r];
                    if (residual != nullptr) v += residual[idx];
                    C[idx] = v;
                }
            }
        }
    }
}

// ---------------------------------------------------------------------------
// Causal depthwise conv1d (width 4) + bias + SiLU. Reads u-part of xz.
// Writes f32 (for scan) and bf16 (for x_proj GEMM).
// ---------------------------------------------------------------------------
__global__ void conv_silu_kernel(const float* __restrict__ xz,
                                 const float* __restrict__ cw,
                                 const float* __restrict__ cb,
                                 float* __restrict__ u,
                                 bf16_t* __restrict__ ubf) {
    int gid = blockIdx.x * blockDim.x + threadIdx.x;
    if (gid >= MTOK * D_INNER) return;
    int m = gid / D_INNER, d = gid % D_INNER;
    int b = m / SEQ, l = m % SEQ;
    float acc = cb[d];
    #pragma unroll
    for (int j = 0; j < D_CONV; ++j) {
        int ll = l - (D_CONV - 1) + j;
        if (ll >= 0)
            acc += xz[(size_t)(b * SEQ + ll) * (2 * D_INNER) + d] * cw[d * D_CONV + j];
    }
    float s = silu_f(acc);
    u[gid]   = s;
    ubf[gid] = f2bf(s);
}

// ---------------------------------------------------------------------------
// delta = softplus(dt @ dt_w^T + dt_b); dt = x_dbl[:, :48] (row stride 80)
// ---------------------------------------------------------------------------
__global__ void dtproj_kernel(const float* __restrict__ xdbl,
                              const float* __restrict__ dtw,
                              const float* __restrict__ dtb,
                              float* __restrict__ delta) {
    int gid = blockIdx.x * blockDim.x + threadIdx.x;
    if (gid >= MTOK * D_INNER) return;
    int m = gid / D_INNER, d = gid % D_INNER;
    const float* row  = xdbl + (size_t)m * XDBL_W;
    const float* wrow = dtw + (size_t)d * DT_RANK;
    float acc = dtb[d];
    #pragma unroll 8
    for (int k = 0; k < DT_RANK; ++k) acc += row[k] * wrow[k];
    delta[gid] = softplus_f(acc);
}

// ---------------------------------------------------------------------------
// Selective scan: one thread per (b, d) channel; 16-state recurrence in regs.
// B/C for each timestep staged into LDS (32 floats), shared by the block.
// ---------------------------------------------------------------------------
__global__ __launch_bounds__(256)
void scan_kernel(const float* __restrict__ delta,
                 const float* __restrict__ u,
                 const float* __restrict__ xdbl,
                 const float* __restrict__ a_log,
                 float* __restrict__ ys) {
    int d = blockIdx.x * 256 + threadIdx.x;   // 0..1535
    int b = blockIdx.y;                       // 0..1
    float A[D_STATE], h[D_STATE];
    #pragma unroll
    for (int n = 0; n < D_STATE; ++n) {
        A[n] = -__expf(a_log[(size_t)d * D_STATE + n]);
        h[n] = 0.0f;
    }
    __shared__ float sBC[2 * D_STATE];
    for (int l = 0; l < SEQ; ++l) {
        int row = b * SEQ + l;
        if (threadIdx.x < 2 * D_STATE)
            sBC[threadIdx.x] = xdbl[(size_t)row * XDBL_W + DT_RANK + threadIdx.x];
        __syncthreads();
        float dlt = delta[(size_t)row * D_INNER + d];
        float uu  = u[(size_t)row * D_INNER + d];
        float y = 0.0f;
        #pragma unroll
        for (int n = 0; n < D_STATE; ++n) {
            float dA = __expf(dlt * A[n]);
            h[n] = dA * h[n] + dlt * sBC[n] * uu;
            y += h[n] * sBC[D_STATE + n];
        }
        ys[(size_t)row * D_INNER + d] = y;
        __syncthreads();
    }
}

// ---------------------------------------------------------------------------
// y = (ys + u * D) * silu(z) -> bf16 (for out_proj GEMM)
// ---------------------------------------------------------------------------
__global__ void gate_kernel(const float* __restrict__ ys,
                            const float* __restrict__ u,
                            const float* __restrict__ xz,
                            const float* __restrict__ Dv,
                            bf16_t* __restrict__ ybf) {
    int gid = blockIdx.x * blockDim.x + threadIdx.x;
    if (gid >= MTOK * D_INNER) return;
    int m = gid / D_INNER, d = gid % D_INNER;
    float z = xz[(size_t)m * (2 * D_INNER) + D_INNER + d];
    float y = (ys[gid] + u[gid] * Dv[d]) * silu_f(z);
    ybf[gid] = f2bf(y);
}

// ---------------------------------------------------------------------------
// Host-side orchestration
// ---------------------------------------------------------------------------
extern "C" void kernel_launch(void* const* d_in, const int* in_sizes, int n_in,
                              void* d_out, int out_size, void* d_ws, size_t ws_size,
                              hipStream_t stream) {
    (void)in_sizes; (void)n_in; (void)out_size; (void)ws_size;

    const int*   tokens  = (const int*)d_in[0];
    const float* emb     = (const float*)d_in[1];
    const float* norm_w  = (const float*)d_in[2];
    const float* in_w    = (const float*)d_in[3];
    const float* conv_w  = (const float*)d_in[4];
    const float* conv_b  = (const float*)d_in[5];
    const float* xp_w    = (const float*)d_in[6];
    const float* dt_w    = (const float*)d_in[7];
    const float* dt_b    = (const float*)d_in[8];
    const float* a_log   = (const float*)d_in[9];
    const float* Dv      = (const float*)d_in[10];
    const float* out_w   = (const float*)d_in[11];
    const float* normf_w = (const float*)d_in[12];
    const float* lm_w    = (const float*)d_in[13];
    float* out = (float*)d_out;

    // Workspace carve-up (~163 MB total)
    char* p = (char*)d_ws;
    auto carve = [&](size_t bytes) -> char* {
        char* r = p;
        p += (bytes + 255) & ~(size_t)255;
        return r;
    };
    float*  x     = (float*) carve((size_t)MTOK * D_MODEL * 4);
    bf16_t* xn    = (bf16_t*)carve((size_t)MTOK * D_MODEL * 2);
    float*  xz    = (float*) carve((size_t)MTOK * 2 * D_INNER * 4);
    float*  u     = (float*) carve((size_t)MTOK * D_INNER * 4);
    bf16_t* ubf   = (bf16_t*)carve((size_t)MTOK * D_INNER * 2);
    float*  xdbl  = (float*) carve((size_t)MTOK * XDBL_W * 4);
    float*  delta = (float*) carve((size_t)MTOK * D_INNER * 4);
    float*  ys    = (float*) carve((size_t)MTOK * D_INNER * 4);
    bf16_t* ybf   = (bf16_t*)carve((size_t)MTOK * D_INNER * 2);
    bf16_t* wbuf  = (bf16_t*)carve((size_t)VOCAB * D_MODEL * 2);  // reused per-GEMM

    embed_kernel<<<(MTOK * D_MODEL + 255) / 256, 256, 0, stream>>>(tokens, emb, x);

    for (int i = 0; i < N_LAYERS; ++i) {
        rmsnorm_bf16_kernel<<<MTOK, 256, 0, stream>>>(x, norm_w + i * D_MODEL, xn, D_MODEL);

        // in_proj: (M,768) x (3072,768)^T -> xz
        int nw = 2 * D_INNER * D_MODEL;
        cvt_bf16_kernel<<<(nw + 255) / 256, 256, 0, stream>>>(in_w + (size_t)i * nw, wbuf, nw);
        gemm_bf16_wmma_kernel<<<dim3((2 * D_INNER + 63) / 64, MTOK / 128), 128, 0, stream>>>(
            xn, wbuf, xz, MTOK, 2 * D_INNER, D_MODEL, nullptr);

        conv_silu_kernel<<<(MTOK * D_INNER + 255) / 256, 256, 0, stream>>>(
            xz, conv_w + (size_t)i * D_INNER * D_CONV, conv_b + i * D_INNER, u, ubf);

        // x_proj: (M,1536) x (80,1536)^T -> x_dbl
        int nx = XDBL_W * D_INNER;
        cvt_bf16_kernel<<<(nx + 255) / 256, 256, 0, stream>>>(xp_w + (size_t)i * nx, wbuf, nx);
        gemm_bf16_wmma_kernel<<<dim3((XDBL_W + 63) / 64, MTOK / 128), 128, 0, stream>>>(
            ubf, wbuf, xdbl, MTOK, XDBL_W, D_INNER, nullptr);

        dtproj_kernel<<<(MTOK * D_INNER + 255) / 256, 256, 0, stream>>>(
            xdbl, dt_w + (size_t)i * D_INNER * DT_RANK, dt_b + i * D_INNER, delta);

        scan_kernel<<<dim3(D_INNER / 256, BATCH), 256, 0, stream>>>(
            delta, u, xdbl, a_log + (size_t)i * D_INNER * D_STATE, ys);

        gate_kernel<<<(MTOK * D_INNER + 255) / 256, 256, 0, stream>>>(
            ys, u, xz, Dv + i * D_INNER, ybf);

        // out_proj: (M,1536) x (768,1536)^T + residual -> x
        int no = D_MODEL * D_INNER;
        cvt_bf16_kernel<<<(no + 255) / 256, 256, 0, stream>>>(out_w + (size_t)i * no, wbuf, no);
        gemm_bf16_wmma_kernel<<<dim3((D_MODEL + 63) / 64, MTOK / 128), 128, 0, stream>>>(
            ybf, wbuf, x, MTOK, D_MODEL, D_INNER, x);
    }

    rmsnorm_bf16_kernel<<<MTOK, 256, 0, stream>>>(x, normf_w, xn, D_MODEL);

    // lm_head: (M,768) x (50257,768)^T -> logits
    int nl = VOCAB * D_MODEL;
    cvt_bf16_kernel<<<(nl + 255) / 256, 256, 0, stream>>>(lm_w, wbuf, nl);
    gemm_bf16_wmma_kernel<<<dim3((VOCAB + 63) / 64, MTOK / 128), 128, 0, stream>>>(
        xn, wbuf, out, MTOK, VOCAB, D_MODEL, nullptr);
}
